// IdentityResBlock_19189913878683
// MI455X (gfx1250) — matrix operand order repaired and checked by reading, not compiled, not run on hardware
//
#include <hip/hip_runtime.h>

// ---------------------------------------------------------------------------
// Types for CDNA5 WMMA
// ---------------------------------------------------------------------------
typedef __attribute__((ext_vector_type(16))) __bf16 v16bf;
typedef __attribute__((ext_vector_type(8)))  float  v8f;

__device__ __forceinline__ v8f wmma_bf16(v16bf a, v16bf b, v8f c) {
  // (neg_a, A, neg_b, B, c_mod, C, reuse_a, reuse_b)
  return __builtin_amdgcn_wmma_f32_16x16x32_bf16(false, a, false, b, (short)0, c,
                                                 false, false);
}

#define EPS 1e-3f

// ---------------------------------------------------------------------------
// Zero the stats accumulators (run every launch; atomics accumulate into them)
// ---------------------------------------------------------------------------
__global__ void k_zero(float* p, int n) {
  int i = blockIdx.x * blockDim.x + threadIdx.x;
  if (i < n) p[i] = 0.0f;
}

// ---------------------------------------------------------------------------
// Repack a row-major fp32 weight [K, ncols] into bf16 WMMA B-fragment layout:
// dst[((kc*ntiles + nt)*32 + lane)*16 + e] = W[(kc*32 + (lane&16?8:0) + (e&7) + (e&8?16:0))*ncols
//                                              + nt*16 + (lane&15)]
// so each lane loads its whole 32x16 B fragment with one 32-byte vector load.
// ---------------------------------------------------------------------------
__global__ void k_prep(const float* __restrict__ src, __bf16* __restrict__ dst,
                       int ntiles, int ncols, int total) {
  int idx = blockIdx.x * blockDim.x + threadIdx.x;
  if (idx >= total) return;
  int e    = idx & 15;
  int lane = (idx >> 4) & 31;
  int nt   = (idx >> 9) % ntiles;
  int kc   = idx / (512 * ntiles);
  int K    = kc * 32 + ((lane & 16) ? 8 : 0) + (e & 7) + ((e & 8) ? 16 : 0);
  int col  = nt * 16 + (lane & 15);
  dst[idx] = (__bf16)src[(size_t)K * ncols + col];
}

// ---------------------------------------------------------------------------
// BN finalize: scale[c] = g*rsqrt(var+eps), bias[c] = b - mean*scale
// ---------------------------------------------------------------------------
__global__ void k_fin(const float* __restrict__ sum, const float* __restrict__ sq,
                      const float* __restrict__ g, const float* __restrict__ b,
                      float* __restrict__ scale, float* __restrict__ bias,
                      int C, float invN) {
  int c = blockIdx.x * blockDim.x + threadIdx.x;
  if (c >= C) return;
  float mean = sum[c] * invN;
  float var  = sq[c] * invN - mean * mean;
  float inv  = rsqrtf(var + EPS);
  float sc   = g[c] * inv;
  scale[c] = sc;
  bias[c]  = b[c] - mean * sc;
}

// ---------------------------------------------------------------------------
// Elementwise BN+ReLU, fp32 -> bf16 (64 channels)
// ---------------------------------------------------------------------------
__global__ void k_bnrelu(const float* __restrict__ t, const float* __restrict__ scale,
                         const float* __restrict__ bias, __bf16* __restrict__ o,
                         int total) {
  int i = blockIdx.x * blockDim.x + threadIdx.x;
  if (i >= total) return;
  int c = i & 63;
  float v = fmaxf(t[i] * scale[c] + bias[c], 0.0f);
  o[i] = (__bf16)v;
}

// ---------------------------------------------------------------------------
// Per-wave tail: accumulate per-channel sum / sum-of-squares of a 16x16 C tile
// into global stats (2 atomics per wave per 16-channel tile).
// ---------------------------------------------------------------------------
__device__ __forceinline__ void stat_tail(const v8f& acc, int lane, float* sum,
                                          float* sq, int cbase) {
  float s = 0.0f, q = 0.0f;
#pragma unroll
  for (int v = 0; v < 8; ++v) {
    float f = acc[v];
    s += f;
    q += f * f;
  }
  s += __shfl_xor(s, 16, 32);
  q += __shfl_xor(q, 16, 32);
  if (lane < 16) {
    atomicAdd(&sum[cbase + lane], s);
    atomicAdd(&sq[cbase + lane], q);
  }
}

// ---------------------------------------------------------------------------
// GEMM1: t1[N,64] = x[N,256] (->bf16) @ W1f ; accumulate BN1 stats
// One wave = one 16x64 output tile. Block = 8 waves.
// ---------------------------------------------------------------------------
__global__ void k_gemm1(const float* __restrict__ x, const __bf16* __restrict__ W1f,
                        float* __restrict__ t1, float* __restrict__ sum,
                        float* __restrict__ sq, int N) {
  int tile = blockIdx.x * 8 + (threadIdx.x >> 5);
  if (tile * 16 >= N) return;
  int lane = threadIdx.x & 31;
  int half = lane >> 4;
  int mrow = tile * 16 + (lane & 15);

  v8f acc[4];
#pragma unroll
  for (int nt = 0; nt < 4; ++nt) acc[nt] = (v8f){0,0,0,0,0,0,0,0};

#pragma unroll
  for (int kc = 0; kc < 8; ++kc) {
    const float4* ap = (const float4*)(x + (size_t)mrow * 256 + kc * 32 + half * 8);
    float4 p0 = ap[0], p1 = ap[1], p2 = ap[4], p3 = ap[5];
    float fa[16] = {p0.x, p0.y, p0.z, p0.w, p1.x, p1.y, p1.z, p1.w,
                    p2.x, p2.y, p2.z, p2.w, p3.x, p3.y, p3.z, p3.w};
    v16bf a;
#pragma unroll
    for (int e = 0; e < 16; ++e) a[e] = (__bf16)fa[e];
#pragma unroll
    for (int nt = 0; nt < 4; ++nt) {
      v16bf b = *(const v16bf*)(W1f + ((size_t)(kc * 4 + nt) * 32 + lane) * 16);
      acc[nt] = wmma_bf16(a, b, acc[nt]);
    }
  }

#pragma unroll
  for (int nt = 0; nt < 4; ++nt) {
    int col = nt * 16 + (lane & 15);
#pragma unroll
    for (int v = 0; v < 8; ++v) {
      int row = tile * 16 + v + half * 8;
      t1[(size_t)row * 64 + col] = acc[nt][v];
    }
    stat_tail(acc[nt], lane, sum, sq, nt * 16);
  }
}

// ---------------------------------------------------------------------------
// GEMM2 (27-neighbor gathered conv):
// h2[n,d] = sum_k sum_c h1[neigh[n,k],c] * W2[k,c,d]; K-dim = 27*64 = 1728.
// h1 is bf16 [N,64] (25.6 MB -> L2-resident gather). Accumulate BN2 stats.
// ---------------------------------------------------------------------------
__global__ void k_gemm2(const __bf16* __restrict__ h1, const int* __restrict__ neigh,
                        const __bf16* __restrict__ W2f, float* __restrict__ h2,
                        float* __restrict__ sum, float* __restrict__ sq, int N) {
  int tile = blockIdx.x * 8 + (threadIdx.x >> 5);
  if (tile * 16 >= N) return;
  int lane = threadIdx.x & 31;
  int half = lane >> 4;
  int mrow = tile * 16 + (lane & 15);
  const int* nrow = neigh + (size_t)mrow * 27;

  v8f acc[4];
#pragma unroll
  for (int nt = 0; nt < 4; ++nt) acc[nt] = (v8f){0,0,0,0,0,0,0,0};

  for (int kc = 0; kc < 54; ++kc) {
    int nb = kc >> 1;
    int cb = (kc & 1) * 32;
    int nr = nrow[nb];
    const __bf16* ap = h1 + (size_t)nr * 64 + cb + half * 8;
    union { v16bf v; uint4 q[2]; } au;
    au.q[0] = *(const uint4*)ap;
    au.q[1] = *(const uint4*)(ap + 16);
    v16bf a = au.v;
#pragma unroll
    for (int nt = 0; nt < 4; ++nt) {
      v16bf b = *(const v16bf*)(W2f + ((size_t)(kc * 4 + nt) * 32 + lane) * 16);
      acc[nt] = wmma_bf16(a, b, acc[nt]);
    }
  }

#pragma unroll
  for (int nt = 0; nt < 4; ++nt) {
    int col = nt * 16 + (lane & 15);
#pragma unroll
    for (int v = 0; v < 8; ++v) {
      int row = tile * 16 + v + half * 8;
      h2[(size_t)row * 64 + col] = acc[nt][v];
    }
    stat_tail(acc[nt], lane, sum, sq, nt * 16);
  }
}

// ---------------------------------------------------------------------------
// GEMM3: t3[N,256] = h2bf[N,64] @ W3f.
// mode 0: accumulate BN3 stats only (t3 never stored -> saves 410 MB HBM).
// mode 1: recompute t3, fuse BN3 + residual(x) + ReLU, store out fp32.
// ---------------------------------------------------------------------------
__global__ void k_gemm3(const __bf16* __restrict__ h2, const __bf16* __restrict__ W3f,
                        const float* __restrict__ x, float* __restrict__ sum,
                        float* __restrict__ sq, const float* __restrict__ scale,
                        const float* __restrict__ bias, float* __restrict__ out,
                        int N, int mode) {
  int tile = blockIdx.x * 8 + (threadIdx.x >> 5);
  if (tile * 16 >= N) return;
  int lane = threadIdx.x & 31;
  int half = lane >> 4;
  int mrow = tile * 16 + (lane & 15);

  v8f acc[16];
#pragma unroll
  for (int nt = 0; nt < 16; ++nt) acc[nt] = (v8f){0,0,0,0,0,0,0,0};

#pragma unroll
  for (int kc = 0; kc < 2; ++kc) {
    const __bf16* ap = h2 + (size_t)mrow * 64 + kc * 32 + half * 8;
    union { v16bf v; uint4 q[2]; } au;
    au.q[0] = *(const uint4*)ap;
    au.q[1] = *(const uint4*)(ap + 16);
    v16bf a = au.v;
#pragma unroll
    for (int nt = 0; nt < 16; ++nt) {
      v16bf b = *(const v16bf*)(W3f + ((size_t)(kc * 16 + nt) * 32 + lane) * 16);
      acc[nt] = wmma_bf16(a, b, acc[nt]);
    }
  }

  if (mode == 0) {
#pragma unroll
    for (int nt = 0; nt < 16; ++nt) stat_tail(acc[nt], lane, sum, sq, nt * 16);
  } else {
#pragma unroll
    for (int nt = 0; nt < 16; ++nt) {
      int c = nt * 16 + (lane & 15);
      float sc = scale[c];
      float bi = bias[c];
#pragma unroll
      for (int v = 0; v < 8; ++v) {
        int row = tile * 16 + v + half * 8;
        float r = acc[nt][v] * sc + bi + x[(size_t)row * 256 + c];
        out[(size_t)row * 256 + c] = fmaxf(r, 0.0f);
      }
    }
  }
}

// ---------------------------------------------------------------------------
// Launch
// ---------------------------------------------------------------------------
extern "C" void kernel_launch(void* const* d_in, const int* in_sizes, int n_in,
                              void* d_out, int out_size, void* d_ws, size_t ws_size,
                              hipStream_t stream) {
  const float* x     = (const float*)d_in[0];
  const int*   neigh = (const int*)d_in[1];
  const float* W1 = (const float*)d_in[3];
  const float* g1 = (const float*)d_in[4];
  const float* b1 = (const float*)d_in[5];
  const float* W2 = (const float*)d_in[6];
  const float* g2 = (const float*)d_in[7];
  const float* b2 = (const float*)d_in[8];
  const float* W3 = (const float*)d_in[9];
  const float* g3 = (const float*)d_in[10];
  const float* b3 = (const float*)d_in[11];

  const int N = in_sizes[0] / 256;
  float* out = (float*)d_out;

  // ---- workspace layout ----
  float* stats = (float*)d_ws;          // 2048 floats (8 KB)
  float* sum1 = stats + 0,    *sq1 = stats + 64;
  float* sum2 = stats + 128,  *sq2 = stats + 192;
  float* sum3 = stats + 256,  *sq3 = stats + 512;
  float* sc1  = stats + 768,  *bi1 = stats + 832;
  float* sc2  = stats + 896,  *bi2 = stats + 960;
  float* sc3  = stats + 1024, *bi3 = stats + 1280;

  __bf16* W1f = (__bf16*)(stats + 2048);     //  8*4*512  = 16384
  __bf16* W2f = W1f + 16384;                 // 54*4*512  = 110592
  __bf16* W3f = W2f + 110592;                //  2*16*512 = 16384
  float*  bufA = (float*)((char*)d_ws + 8192 + (size_t)143360 * 2); // N*64 f32
  __bf16* h1b = (__bf16*)(bufA + (size_t)N * 64);                   // N*64 bf16
  __bf16* h2b = h1b + (size_t)N * 64;                               // N*64 bf16

  const int tiles = (N + 15) / 16;
  const int tb    = (tiles + 7) / 8;
  const int elems = N * 64;
  const float invN = 1.0f / (float)N;

  k_zero<<<3, 256, 0, stream>>>(stats, 768);
  k_prep<<<(16384 + 255) / 256, 256, 0, stream>>>(W1, W1f, 4, 64, 16384);
  k_prep<<<(110592 + 255) / 256, 256, 0, stream>>>(W2, W2f, 4, 64, 110592);
  k_prep<<<(16384 + 255) / 256, 256, 0, stream>>>(W3, W3f, 16, 256, 16384);

  // conv1 + BN1 + ReLU
  k_gemm1<<<tb, 256, 0, stream>>>(x, W1f, bufA, sum1, sq1, N);
  k_fin<<<1, 64, 0, stream>>>(sum1, sq1, g1, b1, sc1, bi1, 64, invN);
  k_bnrelu<<<(elems + 255) / 256, 256, 0, stream>>>(bufA, sc1, bi1, h1b, elems);

  // conv2 (27-neighbor gather) + BN2 + ReLU
  k_gemm2<<<tb, 256, 0, stream>>>(h1b, neigh, W2f, bufA, sum2, sq2, N);
  k_fin<<<1, 64, 0, stream>>>(sum2, sq2, g2, b2, sc2, bi2, 64, invN);
  k_bnrelu<<<(elems + 255) / 256, 256, 0, stream>>>(bufA, sc2, bi2, h2b, elems);

  // conv3: stats pass, finalize, then fused recompute + BN3 + residual + ReLU
  k_gemm3<<<tb, 256, 0, stream>>>(h2b, W3f, x, sum3, sq3, sc3, bi3, out, N, 0);
  k_fin<<<1, 256, 0, stream>>>(sum3, sq3, g3, b3, sc3, bi3, 256, invN);
  k_gemm3<<<tb, 256, 0, stream>>>(h2b, W3f, x, sum3, sq3, sc3, bi3, out, N, 1);
}
